// MoveDecoder_292057776519
// MI455X (gfx1250) — matrix operand order repaired and checked by reading, not compiled, not run on hardware
//
#include <hip/hip_runtime.h>

// ---------------- problem constants (from reference) ----------------
#define BATCH   2048
#define SEQ_L   10
#define VOCAB   1974
#define D_MODEL 512
#define NHEAD   8
#define DHEAD   64
#define DVHEAD  64
#define DFFN    2048
#define NLAYER  6
#define BOARD_L 70

#define BL_ROWS (BATCH * SEQ_L)       // 20480 token rows
#define BD_ROWS (BATCH * BOARD_L)     // 143360 board rows
#define KVDIM   (NHEAD * (DHEAD + DVHEAD))  // 1024

typedef __attribute__((ext_vector_type(16))) __bf16 v16bf;
typedef __attribute__((ext_vector_type(8)))  float  v8f;
typedef unsigned short ushort_t;

__device__ __forceinline__ unsigned short f2bf_bits(float f) {
  union { float f; unsigned u; } x; x.f = f;
  unsigned r = x.u + 0x7FFFu + ((x.u >> 16) & 1u);  // round-to-nearest-even
  return (unsigned short)(r >> 16);
}
__device__ __forceinline__ unsigned pack_bf2(float a, float b) {
  return (unsigned)f2bf_bits(a) | ((unsigned)f2bf_bits(b) << 16);
}
__device__ __forceinline__ float bf2f_lo(unsigned u) {   // low bf16 -> f32 (exact)
  return __builtin_bit_cast(float, u << 16);
}
__device__ __forceinline__ float bf2f_hi(unsigned u) {   // high bf16 -> f32 (exact)
  return __builtin_bit_cast(float, u & 0xFFFF0000u);
}

struct FragBits { uint4 lo, hi; };   // 32 bytes == v16bf

// ---------------------------------------------------------------------------
// GEMM: C[M,N] = A[M,K] @ Wt[N,K]^T (+ bias) (+ relu) (+ residual)
// A and Wt are bf16 (pre-converted); accumulate fp32 via
// v_wmma_f32_16x16x32_bf16.  Block tile 128x128, K-step 32, 8 waves/block,
// each wave owns a 64x32 quadrant (4x2 WMMA tiles).
// Staging: uniform across all 256 threads -- each thread copies a 32B chunk
// of one A row AND one B row (4 clause-able global b128 -> 4 ds_store_b128).
// K-stride 40 halves (80B, 16B aligned) => every fragment is 2x ds_load_b128
// and the 16 lanes hit distinct banks.
// Requires: M % 128 == 0, K % 32 == 0.  N guarded (vocab = 1974).
// ---------------------------------------------------------------------------
#define BM 128
#define BN 128
#define BK 32
#define KLD 40   // padded K stride in halves: 80B rows

__global__ __launch_bounds__(256)
void gemm_bf16_kernel(const ushort_t* __restrict__ A, const ushort_t* __restrict__ Wt,
                      const float* __restrict__ bias, const float* __restrict__ resid,
                      void* __restrict__ Cout, int M, int N, int K, int relu, int bf16out)
{
  __shared__ __align__(16) ushort_t As[BM * KLD];   // 10240 B
  __shared__ __align__(16) ushort_t Bs[BN * KLD];   // 10240 B

  const int tid  = threadIdx.x;
  const int lane = tid & 31;
  const int wave = tid >> 5;            // 0..7
  const int wm   = (wave >> 2) * 64;    // 2 wave-rows
  const int wn   = (wave & 3) * 32;     // 4 wave-cols

  const int rowBase = blockIdx.y * BM;
  const int colBase = blockIdx.x * BN;
  const bool fullN  = (colBase + BN) <= N;

  const int ml     = lane & 15;
  const int hiHalf = lane >> 4;

  v8f acc[4][2];
  #pragma unroll
  for (int i = 0; i < 4; ++i)
    #pragma unroll
    for (int j = 0; j < 2; ++j)
      acc[i][j] = v8f{0.f,0.f,0.f,0.f,0.f,0.f,0.f,0.f};

  // staging: row = tid>>1 (0..127), chunk = (tid&1)*16 halves (32B)
  const int  sra = tid >> 1;
  const int  sch = (tid & 1) * 16;
  const int  nB  = colBase + sra;
  const bool bvalid = nB < N;

  const ushort_t* aptr = A  + (size_t)(rowBase + sra) * K + sch;
  const ushort_t* wptr = Wt + (size_t)(bvalid ? nB : 0) * K + sch;
  const int sAs = sra * KLD + sch;      // LDS half-index (16B aligned)

  for (int kk = 0; kk < K; kk += BK) {
    // gather all 4 chunks into registers first (clause + single wait)
    const uint4* ap = (const uint4*)(aptr + kk);
    const uint4* wp = (const uint4*)(wptr + kk);
    uint4 a0 = ap[0], a1 = ap[1];
    uint4 w0 = wp[0], w1 = wp[1];
    __builtin_prefetch((const char*)(aptr + kk + BK), 0, 1);
    __builtin_prefetch((const char*)(wptr + kk + BK), 0, 1);
    if (!bvalid) {
      w0 = uint4{0u,0u,0u,0u};
      w1 = uint4{0u,0u,0u,0u};
    }
    *(uint4*)&As[sAs]     = a0;
    *(uint4*)&As[sAs + 8] = a1;
    *(uint4*)&Bs[sAs]     = w0;
    *(uint4*)&Bs[sAs + 8] = w1;
    __syncthreads();

    // ---- fragments: 2x ds_load_b128 each ----
    v16bf afrag[4], bfrag[2];
    #pragma unroll
    for (int mt = 0; mt < 4; ++mt) {
      const int base = (wm + mt * 16 + ml) * KLD + hiHalf * 8;
      FragBits fr;
      fr.lo = *(const uint4*)&As[base];        // K = hiHalf*8 + 0..7
      fr.hi = *(const uint4*)&As[base + 16];   // K = 16 + hiHalf*8 + 0..7
      afrag[mt] = __builtin_bit_cast(v16bf, fr);
    }
    #pragma unroll
    for (int nt = 0; nt < 2; ++nt) {
      const int base = (wn + nt * 16 + ml) * KLD + hiHalf * 8;
      FragBits fr;
      fr.lo = *(const uint4*)&Bs[base];        // K = hiHalf*8 + 0..7
      fr.hi = *(const uint4*)&Bs[base + 16];   // K = 16 + hiHalf*8 + 0..7
      bfrag[nt] = __builtin_bit_cast(v16bf, fr);
    }

    #pragma unroll
    for (int mt = 0; mt < 4; ++mt)
      #pragma unroll
      for (int nt = 0; nt < 2; ++nt)
        acc[mt][nt] = __builtin_amdgcn_wmma_f32_16x16x32_bf16(
            false, afrag[mt], false, bfrag[nt], (short)0, acc[mt][nt], false, false);

    __syncthreads();
  }

  // ---- epilogue: VGPR r holds M = r + 8*hiHalf, N = lane%16 ----
  float*    Cf = (float*)Cout;
  ushort_t* Cb = (ushort_t*)Cout;
  #pragma unroll
  for (int nt = 0; nt < 2; ++nt) {
    const int n = colBase + wn + nt * 16 + ml;
    if (fullN || n < N) {
      const float bv = bias ? bias[n] : 0.0f;
      #pragma unroll
      for (int mt = 0; mt < 4; ++mt)
        #pragma unroll
        for (int r = 0; r < 8; ++r) {
          const int m = rowBase + wm + mt * 16 + hiHalf * 8 + r;
          float val = acc[mt][nt][r] + bv;
          if (relu) val = fmaxf(val, 0.0f);
          if (bf16out) {
            Cb[(size_t)m * N + n] = f2bf_bits(val);
          } else {
            if (resid) val += resid[(size_t)m * N + n];
            Cf[(size_t)m * N + n] = val;
          }
        }
    }
  }
}

// ---------------------------------------------------------------------------
// Weight convert+transpose: W[K][N] fp32 -> Wt[N][K] bf16.  32x32 LDS tiles.
// ---------------------------------------------------------------------------
__global__ __launch_bounds__(256)
void wconvert_kernel(const float* __restrict__ W, ushort_t* __restrict__ Wt,
                     int K, int N)
{
  __shared__ float t[32][33];
  const int tx = threadIdx.x & 31;
  const int ty = threadIdx.x >> 5;      // 0..7
  const int k0 = blockIdx.y * 32;
  const int n0 = blockIdx.x * 32;
  #pragma unroll
  for (int i = 0; i < 4; ++i) {
    const int k = k0 + ty + i * 8;
    const int n = n0 + tx;
    t[ty + i * 8][tx] = (n < N) ? W[(size_t)k * N + n] : 0.0f;
  }
  __syncthreads();
  #pragma unroll
  for (int i = 0; i < 4; ++i) {
    const int n = n0 + ty + i * 8;
    const int k = k0 + tx;
    if (n < N) Wt[(size_t)n * K + k] = f2bf_bits(t[tx][ty + i * 8]);
  }
}

// ---------------------------------------------------------------------------
// Elementwise fp32 -> bf16 convert (for boards), vectorized 4-wide.
// Count must be a multiple of 1024 (it is: BD_ROWS*512).
// ---------------------------------------------------------------------------
__global__ __launch_bounds__(256)
void bconvert_kernel(const float* __restrict__ src, ushort_t* __restrict__ dst)
{
  const size_t i = (size_t)blockIdx.x * 256 + threadIdx.x;
  const float4 f = ((const float4*)src)[i];
  uint2 u; u.x = pack_bf2(f.x, f.y); u.y = pack_bf2(f.z, f.w);
  ((uint2*)dst)[i] = u;
}

// ---------------------------------------------------------------------------
// LayerNorm over 512-wide rows -> bf16 output.  One wave per row; each lane
// owns 16 consecutive columns (4x float4 in, 4x b64 out).
// ---------------------------------------------------------------------------
__global__ __launch_bounds__(256)
void layernorm_kernel(const float* __restrict__ X, const float* __restrict__ g,
                      const float* __restrict__ bta, ushort_t* __restrict__ Y, int rows)
{
  const int wave = threadIdx.x >> 5;
  const int lane = threadIdx.x & 31;
  const int row  = blockIdx.x * 8 + wave;
  if (row >= rows) return;
  const int c0 = lane * 16;
  const float4* x4 = (const float4*)(X + (size_t)row * D_MODEL + c0);
  float4 f[4];
  float s = 0.f;
  #pragma unroll
  for (int j = 0; j < 4; ++j) {
    f[j] = x4[j];
    s += f[j].x + f[j].y + f[j].z + f[j].w;
  }
  #pragma unroll
  for (int o = 16; o > 0; o >>= 1) s += __shfl_xor(s, o, 32);
  const float mean = s * (1.0f / 512.0f);
  float q = 0.f;
  #pragma unroll
  for (int j = 0; j < 4; ++j) {
    float dx = f[j].x - mean, dy = f[j].y - mean, dz = f[j].z - mean, dw = f[j].w - mean;
    q += dx * dx + dy * dy + dz * dz + dw * dw;
  }
  #pragma unroll
  for (int o = 16; o > 0; o >>= 1) q += __shfl_xor(q, o, 32);
  const float rinv = rsqrtf(q * (1.0f / 512.0f) + 1e-5f);

  const float4* g4 = (const float4*)(g + c0);
  const float4* b4 = (const float4*)(bta + c0);
  uint2* y2 = (uint2*)(Y + (size_t)row * D_MODEL + c0);
  #pragma unroll
  for (int j = 0; j < 4; ++j) {
    const float4 gv = g4[j], bv = b4[j];
    uint2 u;
    u.x = pack_bf2((f[j].x - mean) * rinv * gv.x + bv.x,
                   (f[j].y - mean) * rinv * gv.y + bv.y);
    u.y = pack_bf2((f[j].z - mean) * rinv * gv.z + bv.z,
                   (f[j].w - mean) * rinv * gv.w + bv.w);
    y2[j] = u;
  }
}

// ---------------------------------------------------------------------------
// Embedding: x = emb[moves]*sqrt(D) + pos   (fp32 residual stream)
// ---------------------------------------------------------------------------
__global__ __launch_bounds__(256)
void embed_kernel(const int* __restrict__ moves, const float* __restrict__ emb,
                  const float* __restrict__ pos, float* __restrict__ X)
{
  const int row = blockIdx.x;
  const int l   = row % SEQ_L;
  const int tok = moves[row];
  const float scale = 22.62741699796952f;     // sqrt(512)
  for (int c = threadIdx.x; c < D_MODEL; c += 256)
    X[(size_t)row * D_MODEL + c] =
        emb[(size_t)tok * D_MODEL + c] * scale + pos[l * D_MODEL + c];
}

// ---------------------------------------------------------------------------
// Attention (bf16 in / bf16 out): one wave per (b, head, query row).
// Online softmax; fp32 math; bf16->f32 is an exact <<16.
// ---------------------------------------------------------------------------
__global__ __launch_bounds__(32)
void attn_kernel(const ushort_t* __restrict__ Q, const ushort_t* __restrict__ KV,
                 ushort_t* __restrict__ O, const int* __restrict__ lengths,
                 int Lq, int Lk, int causal)
{
  const int idx  = blockIdx.x;
  const int qi   = idx % Lq;
  const int h    = (idx / Lq) % NHEAD;
  const int b    = idx / (Lq * NHEAD);
  const int lane = threadIdx.x;

  const unsigned qp = *(const unsigned*)(Q + (size_t)(b * Lq + qi) * D_MODEL + h * DHEAD + lane * 2);
  const float q0 = bf2f_lo(qp), q1 = bf2f_hi(qp);

  int len = causal ? (qi + 1) : Lk;
  if (lengths) { int lb = lengths[b]; len = len < lb ? len : lb; }

  float m = -3.4e38f, l = 0.f, a0 = 0.f, a1 = 0.f;
  for (int kj = 0; kj < len; ++kj) {
    const size_t kvrow = (size_t)(b * Lk + kj) * KVDIM;
    const unsigned kp = *(const unsigned*)(KV + kvrow + h * DHEAD + lane * 2);
    float p = q0 * bf2f_lo(kp) + q1 * bf2f_hi(kp);
    #pragma unroll
    for (int o = 16; o > 0; o >>= 1) p += __shfl_xor(p, o, 32);
    p *= 0.125f;                           // 1/sqrt(64)
    const float mn = fmaxf(m, p);
    const float sc = __expf(m - mn);
    const float e  = __expf(p - mn);
    const unsigned vp = *(const unsigned*)(KV + kvrow + (NHEAD * DHEAD) + h * DVHEAD + lane * 2);
    l  = l  * sc + e;
    a0 = a0 * sc + e * bf2f_lo(vp);
    a1 = a1 * sc + e * bf2f_hi(vp);
    m = mn;
  }
  const float inv = 1.0f / l;
  *(unsigned*)(O + (size_t)(b * Lq + qi) * D_MODEL + h * DVHEAD + lane * 2) =
      pack_bf2(a0 * inv, a1 * inv);
}

// ---------------------------------------------------------------------------
// Host orchestration
// ---------------------------------------------------------------------------
extern "C" void kernel_launch(void* const* d_in, const int* in_sizes, int n_in,
                              void* d_out, int out_size, void* d_ws, size_t ws_size,
                              hipStream_t stream) {
  (void)in_sizes; (void)n_in; (void)out_size; (void)ws_size;

  const int*   moves   = (const int*)  d_in[0];
  const int*   lengths = (const int*)  d_in[1];
  const float* boards  = (const float*)d_in[2];
  const float* emb     = (const float*)d_in[3];
  const float* pos     = (const float*)d_in[4];
  const float* sa_ln_g = (const float*)d_in[5];
  const float* sa_ln_b = (const float*)d_in[6];
  const float* sa_wq   = (const float*)d_in[7];
  const float* sa_bq   = (const float*)d_in[8];
  const float* sa_wkv  = (const float*)d_in[9];
  const float* sa_bkv  = (const float*)d_in[10];
  const float* sa_wo   = (const float*)d_in[11];
  const float* sa_bo   = (const float*)d_in[12];
  const float* ca_ln_g = (const float*)d_in[13];
  const float* ca_ln_b = (const float*)d_in[14];
  const float* ca_wq   = (const float*)d_in[15];
  const float* ca_bq   = (const float*)d_in[16];
  const float* ca_wkv  = (const float*)d_in[17];
  const float* ca_bkv  = (const float*)d_in[18];
  const float* ca_wo   = (const float*)d_in[19];
  const float* ca_bo   = (const float*)d_in[20];
  const float* ff_ln_g = (const float*)d_in[21];
  const float* ff_ln_b = (const float*)d_in[22];
  const float* ff_w1   = (const float*)d_in[23];
  const float* ff_b1   = (const float*)d_in[24];
  const float* ff_w2   = (const float*)d_in[25];
  const float* ff_b2   = (const float*)d_in[26];
  const float* fin_g   = (const float*)d_in[27];
  const float* fin_b   = (const float*)d_in[28];
  const float* fc_w    = (const float*)d_in[29];
  const float* fc_b    = (const float*)d_in[30];

  // ---- workspace layout ----
  // X (fp32 residual stream) then bf16 tensors (as ushort).
  char* ws = (char*)d_ws;
  float*    X    = (float*)ws;                       ws += (size_t)BL_ROWS * D_MODEL * 4;
  ushort_t* XNb  = (ushort_t*)ws;                    ws += (size_t)BL_ROWS * D_MODEL * 2;
  ushort_t* Qb   = (ushort_t*)ws;                    ws += (size_t)BL_ROWS * D_MODEL * 2;
  ushort_t* Ob   = (ushort_t*)ws;                    ws += (size_t)BL_ROWS * D_MODEL * 2;
  ushort_t* BIGb = (ushort_t*)ws;                    ws += (size_t)BD_ROWS * KVDIM * 2;   // KV / FFN-H
  ushort_t* BRDb = (ushort_t*)ws;                    ws += (size_t)BD_ROWS * D_MODEL * 2; // boards bf16
  ushort_t* WTb  = (ushort_t*)ws;                    // 2MB scratch: largest weight 512x2048

  auto wconv = [&](const float* W, int K, int N) {
    dim3 grid((N + 31) / 32, K / 32);
    wconvert_kernel<<<grid, 256, 0, stream>>>(W, WTb, K, N);
  };
  auto gemm = [&](const ushort_t* A, const float* bias, const float* resid,
                  void* Cout, int M, int N, int K, int relu, int bf16out) {
    dim3 grid((N + BN - 1) / BN, M / BM);
    gemm_bf16_kernel<<<grid, 256, 0, stream>>>(A, WTb, bias, resid, Cout, M, N, K, relu, bf16out);
  };
  auto lnorm = [&](const float* x, const float* g, const float* b, ushort_t* y) {
    layernorm_kernel<<<BL_ROWS / 8, 256, 0, stream>>>(x, g, b, y, BL_ROWS);
  };

  // one-time per forward: boards -> bf16; embedding -> X
  bconvert_kernel<<<(BD_ROWS * D_MODEL / 4) / 256, 256, 0, stream>>>(boards, BRDb);
  embed_kernel<<<BL_ROWS, 256, 0, stream>>>(moves, emb, pos, X);

  for (int il = 0; il < NLAYER; ++il) {
    const size_t wD  = (size_t)il * D_MODEL;
    const size_t wQW = (size_t)il * D_MODEL * D_MODEL;
    const size_t wKW = (size_t)il * D_MODEL * KVDIM;

    // ---- self-attention (pre-LN, causal + length mask, residual) ----
    lnorm(X, sa_ln_g + wD, sa_ln_b + wD, XNb);
    wconv(sa_wq + wQW, D_MODEL, D_MODEL);
    gemm(XNb, sa_bq + (size_t)il * D_MODEL, nullptr, Qb,   BL_ROWS, D_MODEL, D_MODEL, 0, 1);
    wconv(sa_wkv + wKW, D_MODEL, KVDIM);
    gemm(XNb, sa_bkv + (size_t)il * KVDIM,  nullptr, BIGb, BL_ROWS, KVDIM,   D_MODEL, 0, 1);
    attn_kernel<<<BATCH * NHEAD * SEQ_L, 32, 0, stream>>>(Qb, BIGb, Ob, lengths, SEQ_L, SEQ_L, 1);
    wconv(sa_wo + wQW, D_MODEL, D_MODEL);
    gemm(Ob, sa_bo + wD, X, X, BL_ROWS, D_MODEL, D_MODEL, 0, 0);

    // ---- cross-attention over boards (boards NOT layer-normed) ----
    lnorm(X, ca_ln_g + wD, ca_ln_b + wD, XNb);
    wconv(ca_wq + wQW, D_MODEL, D_MODEL);
    gemm(XNb, ca_bq + (size_t)il * D_MODEL, nullptr, Qb,   BL_ROWS, D_MODEL, D_MODEL, 0, 1);
    wconv(ca_wkv + wKW, D_MODEL, KVDIM);
    gemm(BRDb, ca_bkv + (size_t)il * KVDIM, nullptr, BIGb, BD_ROWS, KVDIM,   D_MODEL, 0, 1);
    attn_kernel<<<BATCH * NHEAD * SEQ_L, 32, 0, stream>>>(Qb, BIGb, Ob, nullptr, SEQ_L, BOARD_L, 0);
    wconv(ca_wo + wQW, D_MODEL, D_MODEL);
    gemm(Ob, ca_bo + wD, X, X, BL_ROWS, D_MODEL, D_MODEL, 0, 0);

    // ---- FFN (pre-LN, relu, residual) ----
    lnorm(X, ff_ln_g + wD, ff_ln_b + wD, XNb);
    wconv(ff_w1 + (size_t)il * D_MODEL * DFFN, D_MODEL, DFFN);
    gemm(XNb, ff_b1 + (size_t)il * DFFN, nullptr, BIGb, BL_ROWS, DFFN, D_MODEL, 1, 1);
    wconv(ff_w2 + (size_t)il * DFFN * D_MODEL, DFFN, D_MODEL);
    gemm(BIGb, ff_b2 + wD, X, X, BL_ROWS, D_MODEL, DFFN, 0, 0);
  }

  // final LN + vocab projection -> logits (fp32)
  lnorm(X, fin_g, fin_b, XNb);
  wconv(fc_w, D_MODEL, VOCAB);
  gemm(XNb, fc_b, nullptr, d_out, BL_ROWS, VOCAB, D_MODEL, 0, 0);
}
